// CodeEnhance4321_Arch_16329465659946
// MI455X (gfx1250) — matrix operand order repaired and checked by reading, not compiled, not run on hardware
//
#include <hip/hip_runtime.h>

typedef __attribute__((ext_vector_type(16))) __bf16 v16bf;
typedef __attribute__((ext_vector_type(8)))  __bf16 v8bf;
typedef __attribute__((ext_vector_type(8)))  float  v8f;

#define BATCH   32
#define E_DIM   256
#define HH      32
#define WW      32
#define N_VEC   (BATCH*HH*WW)         // 32768 vectors
#define N_CODE  1024
#define OUT_ZQ  (BATCH*E_DIM*HH*WW)   // 8388608 floats
#define OUT_LOSS_OFF OUT_ZQ
#define OUT_IDX_OFF  (OUT_ZQ + 1)

__device__ __forceinline__ __bf16 f2bf(float f) { return (__bf16)f; }

// ---------------------------------------------------------------------------
// Kernel 1: codebook fp32 -> bf16, plus ||c||^2 per row.
// ---------------------------------------------------------------------------
__global__ void __launch_bounds__(256)
cb_prep_kernel(const float* __restrict__ cb, __bf16* __restrict__ cb16,
               float* __restrict__ cnorm) {
    int row = blockIdx.x;
    int t   = threadIdx.x;
    float v = cb[(size_t)row * E_DIM + t];
    cb16[(size_t)row * E_DIM + t] = f2bf(v);
    __shared__ float s[256];
    s[t] = v * v;
    __syncthreads();
    for (int o = 128; o > 0; o >>= 1) {
        if (t < o) s[t] += s[t + o];
        __syncthreads();
    }
    if (t == 0) cnorm[row] = s[0];
}

// ---------------------------------------------------------------------------
// Kernel 2: z (B,C,H,W) fp32 -> z16 (N, E_DIM) bf16, transposed via LDS tile
// so both the global read (along W) and write (along C) are coalesced.
// One block per (b, h): 32 w-positions x 256 channels.
// ---------------------------------------------------------------------------
__global__ void __launch_bounds__(256)
z_trans_kernel(const float* __restrict__ z, __bf16* __restrict__ z16) {
    __shared__ __bf16 tile[32][E_DIM + 2];   // 4B pad -> conflict-free banks
    int b = blockIdx.x >> 5;
    int h = blockIdx.x & 31;
    int t = threadIdx.x;
    int w  = t & 31;
    int ks = t >> 5;                          // 8 channel slices
    const float* zp = z + (size_t)b * E_DIM * HH * WW + (size_t)h * WW + w;
    for (int k = ks; k < E_DIM; k += 8)
        tile[w][k] = f2bf(zp[(size_t)k * HH * WW]);
    __syncthreads();
    size_t mBase = (size_t)b * HH * WW + (size_t)h * WW;
    for (int e = t; e < 32 * E_DIM; e += 256) {
        int w2 = e >> 8;
        int k2 = e & 255;
        z16[(mBase + w2) * E_DIM + k2] = tile[w2][k2];
    }
}

// ---------------------------------------------------------------------------
// Kernel 3: fused distance-GEMM + argmin via v_wmma_f32_16x16x32_bf16.
// Block = 256 threads = 8 waves; each wave owns a 16-row strip of z and loops
// all 64 codebook column tiles, keeping A fragments resident in VGPRs.
// dist(m,n) = ||c_n||^2 - 2 * z_m . c_n   (||z||^2 dropped: constant per row)
//
// Fragment layouts (wave32, 16-bit data, contiguous K as used here, not the ISA table order):
//   A (16x32):  lanes 0-15 = row r, halves 0..7 <- K[0..7], 8..15 <- K[8..15]
//               lanes 16-31 = row r, K offset +16 (K[16..23], K[24..31])
//   B (32x16):  lanes 0-15 = col n, V0..V7 <- K[0..15] (2/VGPR, contiguous)
//               lanes 16-31 = col n, K[16..31]
//   C (16x16):  lane holds col N = lane&15, rows M = j + 8*(lane>>4)
// ---------------------------------------------------------------------------
__global__ void __launch_bounds__(256)
vq_argmin_kernel(const __bf16* __restrict__ z16, const __bf16* __restrict__ cb16,
                 const float* __restrict__ cnorm, int* __restrict__ idx) {
    int lane = threadIdx.x & 31;
    int wave = threadIdx.x >> 5;
    int r    = lane & 15;
    int hsel = lane >> 4;
    int rowBase = (blockIdx.x * 8 + wave) * 16;

    union Frag { v16bf v; v8bf h[2]; };

    // A fragments: 16 rows x 256 K resident for the whole kernel (64 VGPRs).
    Frag a[8];
    const __bf16* ap = z16 + (size_t)(rowBase + r) * E_DIM + hsel * 16;
#pragma unroll
    for (int kc = 0; kc < 8; ++kc) {
        a[kc].h[0] = *(const v8bf*)(ap + kc * 32);
        a[kc].h[1] = *(const v8bf*)(ap + kc * 32 + 8);
    }

    float bestv[8];
    int   besti[8];
#pragma unroll
    for (int j = 0; j < 8; ++j) { bestv[j] = 3.4e38f; besti[j] = 0; }

    for (int ct = 0; ct < 64; ++ct) {
        int col = ct * 16 + r;                       // this lane's B/C column
        // B layout: this lane needs 16 *contiguous* K values per 32-K chunk,
        // starting at K = kc*32 + hsel*16 of codebook row `col`.
        const __bf16* bp = cb16 + (size_t)col * E_DIM + hsel * 16;
        v8f acc = {0.f, 0.f, 0.f, 0.f, 0.f, 0.f, 0.f, 0.f};
#pragma unroll
        for (int kc = 0; kc < 8; ++kc) {
            Frag bf;
            bf.h[0] = *(const v8bf*)(bp + kc * 32);
            bf.h[1] = *(const v8bf*)(bp + kc * 32 + 8);
            acc = __builtin_amdgcn_wmma_f32_16x16x32_bf16(
                false, a[kc].v, false, bf.v, (short)0, acc, false, false);
        }
        float cn = cnorm[col];
#pragma unroll
        for (int j = 0; j < 8; ++j) {
            float d = __builtin_fmaf(-2.0f, acc[j], cn);
            if (d < bestv[j]) { bestv[j] = d; besti[j] = col; }
        }
    }

    // Reduce (min,argmin) across the 16 lanes sharing each row-half;
    // tie-break to the lower codebook index (matches argmin semantics).
#pragma unroll
    for (int j = 0; j < 8; ++j) {
#pragma unroll
        for (int m = 1; m < 16; m <<= 1) {
            float ov = __shfl_xor(bestv[j], m, 32);
            int   oi = __shfl_xor(besti[j], m, 32);
            if (ov < bestv[j] || (ov == bestv[j] && oi < besti[j])) {
                bestv[j] = ov; besti[j] = oi;
            }
        }
    }
    if (r == 0) {
#pragma unroll
        for (int j = 0; j < 8; ++j)
            idx[rowBase + 8 * hsel + j] = besti[j];   // C rows = j + 8*hsel
    }
}

// ---------------------------------------------------------------------------
// Kernel 4: gather z_q = codebook[idx] into (B,C,H,W) layout, emit float idx,
// and accumulate per-block loss partials (deterministic, no atomics).
// One block per (b, h); coalesced over w.
// ---------------------------------------------------------------------------
__global__ void __launch_bounds__(256)
gather_loss_kernel(const float* __restrict__ z, const float* __restrict__ cb,
                   const int* __restrict__ idx, float* __restrict__ out,
                   float* __restrict__ partials) {
    int b = blockIdx.x >> 5;
    int h = blockIdx.x & 31;
    int t = threadIdx.x;
    int w  = t & 31;
    int ks = t >> 5;
    size_t m = (size_t)b * HH * WW + (size_t)h * WW + w;
    int id = idx[m];
    if (ks == 0) out[(size_t)OUT_IDX_OFF + m] = (float)id;
    const float* crow = cb + (size_t)id * E_DIM;
    float sum = 0.0f;
    for (int k = ks; k < E_DIM; k += 8) {
        size_t zoff = (((size_t)b * E_DIM + k) * HH + h) * WW + w;
        float c = crow[k];
        float d = c - z[zoff];
        sum = __builtin_fmaf(d, d, sum);
        out[zoff] = c;                      // straight-through forward == z_q
    }
    __shared__ float s[256];
    s[t] = sum;
    __syncthreads();
    for (int o = 128; o > 0; o >>= 1) {
        if (t < o) s[t] += s[t + o];
        __syncthreads();
    }
    if (t == 0) partials[blockIdx.x] = s[0];
}

// ---------------------------------------------------------------------------
// Kernel 5: fixed-order reduction of 1024 partials -> codebook_loss.
// loss = q + beta*e = 1.25 * mean((z_q - z)^2)
// ---------------------------------------------------------------------------
__global__ void __launch_bounds__(256)
loss_finish_kernel(const float* __restrict__ partials, float* __restrict__ out) {
    __shared__ float s[256];
    int t = threadIdx.x;
    float v = 0.0f;
    for (int i = t; i < BATCH * HH; i += 256) v += partials[i];
    s[t] = v;
    __syncthreads();
    for (int o = 128; o > 0; o >>= 1) {
        if (t < o) s[t] += s[t + o];
        __syncthreads();
    }
    if (t == 0) out[OUT_LOSS_OFF] = s[0] * 1.25f / (float)OUT_ZQ;
}

// ---------------------------------------------------------------------------
// Workspace layout (bytes):
//   [4096,   8192)  : cnorm      (1024 f32)
//   [8192,  12288)  : partials   (1024 f32)
//   [16384, 147456) : idx        (32768 i32)
//   [147456,671744) : cb16       (1024*256 bf16 = 512 KB)
//   [671744, +16MB) : z16        (32768*256 bf16 = 16 MB)
// ---------------------------------------------------------------------------
extern "C" void kernel_launch(void* const* d_in, const int* in_sizes, int n_in,
                              void* d_out, int out_size, void* d_ws, size_t ws_size,
                              hipStream_t stream) {
    (void)in_sizes; (void)n_in; (void)out_size; (void)ws_size;
    const float* z  = (const float*)d_in[0];
    const float* cb = (const float*)d_in[1];
    float* out = (float*)d_out;
    char*  ws  = (char*)d_ws;

    float*  cnorm    = (float*)(ws + 4096);
    float*  partials = (float*)(ws + 8192);
    int*    idxbuf   = (int*)  (ws + 16384);
    __bf16* cb16     = (__bf16*)(ws + 147456);
    __bf16* z16      = (__bf16*)(ws + 671744);

    cb_prep_kernel    <<<N_CODE,      256, 0, stream>>>(cb, cb16, cnorm);
    z_trans_kernel    <<<BATCH * HH,  256, 0, stream>>>(z, z16);
    vq_argmin_kernel  <<<N_VEC / 128, 256, 0, stream>>>(z16, cb16, cnorm, idxbuf);
    gather_loss_kernel<<<BATCH * HH,  256, 0, stream>>>(z, cb, idxbuf, out, partials);
    loss_finish_kernel<<<1,           256, 0, stream>>>(partials, out);
}
